// SelfAttention_6236292514395
// MI455X (gfx1250) — matrix operand order repaired and checked by reading, not compiled, not run on hardware
//
#include <hip/hip_runtime.h>
#include <hip/hip_bf16.h>

typedef __attribute__((ext_vector_type(16))) __bf16 v16bf;
typedef __attribute__((ext_vector_type(8)))  float  v8f;
typedef int v4i __attribute__((vector_size(16)));   // matches async-LDS builtin param
typedef unsigned short u16;
typedef unsigned int   u32;

#define DIM_   3072
#define HEADS_ 24
#define HD_    128
#define NB_    2
#define L_     2048
#define BL_    4096
#define QH_    12582912   // 2*24*2048*128 bf16 elements per q/k/v/ao buffer

#if __has_builtin(__builtin_amdgcn_global_load_async_to_lds_b128) && \
    __has_builtin(__builtin_amdgcn_s_wait_asynccnt)
#define HAVE_ASYNC_LDS 1
#else
#define HAVE_ASYNC_LDS 0
#endif

union FragBF {
  v16bf bf;
  u16   u[16];
};

__device__ __forceinline__ u16 f2bfu(float f) {
  union { float f; u32 u; } a; a.f = f;
  u32 u = a.u;
  u += 0x7FFFu + ((u >> 16) & 1u);   // round-to-nearest-even
  return (u16)(u >> 16);
}

__device__ __forceinline__ u32 pk2bf(float lo, float hi) {
  return (u32)f2bfu(lo) | ((u32)f2bfu(hi) << 16);
}

// 16B global -> LDS copy; async (ASYNCcnt) when the CDNA5 builtin is present.
__device__ __forceinline__ void async_ld16(void* lds, const void* g) {
#if HAVE_ASYNC_LDS
  __builtin_amdgcn_global_load_async_to_lds_b128(
      (__attribute__((address_space(1))) v4i*)(void*)g,
      (__attribute__((address_space(3))) v4i*)lds, 0, 0);
#else
  *(uint4*)lds = *(const uint4*)g;
#endif
}

__device__ __forceinline__ void async_fence() {
#if HAVE_ASYNC_LDS
  __builtin_amdgcn_s_wait_asynccnt(0);
#endif
}

// ---------------------------------------------------------------------------
// Kernel 1: qkv = x @ Wqkv^T + b, fused RMSNorm + RoPE, bf16 outputs [B,H,L,D]
// Tile: 64 tokens x 128 cols (one head-slice of q, k, or v). grid=(64, 72)
// ---------------------------------------------------------------------------
__global__ __launch_bounds__(256) void qkv_fused_kernel(
    const float* __restrict__ x, const float* __restrict__ pe,
    const float* __restrict__ Wqkv, const float* __restrict__ bqkv,
    const float* __restrict__ q_scale, const float* __restrict__ k_scale,
    u16* __restrict__ q_ws, u16* __restrict__ k_ws, u16* __restrict__ v_ws)
{
  __shared__ u16   xs[64][40];      // 64x32 bf16 A tile (+pad)
  __shared__ u16   wsm[128][40];    // 128x32 bf16 B tile (+pad)
  __shared__ float outs[64][132];   // f32 staging for fused epilogue

  const int tid  = threadIdx.x;
  const int lane = tid & 31;
  const int wv   = tid >> 5;
  const int r    = lane & 15;
  const int lh   = lane >> 4;

  const int m0    = blockIdx.x * 64;
  const int nt    = blockIdx.y;            // 0..71
  const int sel   = nt / HEADS_;           // 0=q 1=k 2=v
  const int head  = nt % HEADS_;
  const int nbase = sel * DIM_ + head * HD_;

  const int mi     = wv >> 1;              // 4 M-tiles of 16
  const int njbase = (wv & 1) * 4;         // 8 N-tiles of 16, 4 per wave

  v8f vz = {};
  v8f acc[4];
  for (int t = 0; t < 4; ++t) acc[t] = vz;

  for (int kk = 0; kk < DIM_; kk += 32) {
    // straight-line staging: b128 loads, packed bf16 b64 stores to LDS
#pragma unroll
    for (int it = 0; it < 2; ++it) {               // x: 64x32 = 512 float4
      const int idx = tid + it * 256;
      const int rr = idx >> 3, cc = (idx & 7) * 4;
      float4 v = *(const float4*)(x + (size_t)(m0 + rr) * DIM_ + kk + cc);
      uint2 pk; pk.x = pk2bf(v.x, v.y); pk.y = pk2bf(v.z, v.w);
      *(uint2*)&xs[rr][cc] = pk;
    }
#pragma unroll
    for (int it = 0; it < 4; ++it) {               // W: 128x32 = 1024 float4
      const int idx = tid + it * 256;
      const int rr = idx >> 3, cc = (idx & 7) * 4;
      float4 v = *(const float4*)(Wqkv + (size_t)(nbase + rr) * DIM_ + kk + cc);
      uint2 pk; pk.x = pk2bf(v.x, v.y); pk.y = pk2bf(v.z, v.w);
      *(uint2*)&wsm[rr][cc] = pk;
    }
    __syncthreads();

    FragBF a;
    {
      // A k order used here (same as B): lanes<16 K 0..15, lanes>=16 K 16..31
      const u16* p0 = &xs[mi * 16 + r][lh * 16];
      const u16* p1 = &xs[mi * 16 + r][lh * 16 + 8];
#pragma unroll
      for (int j = 0; j < 8; ++j) { a.u[j] = p0[j]; a.u[8 + j] = p1[j]; }
    }
#pragma unroll
    for (int t = 0; t < 4; ++t) {
      FragBF b;
      // B k order used here (same as A): lanes<16 col=lane K 0..15, lanes>=16 col=lane-16 K 16..31
      const u16* p = &wsm[(njbase + t) * 16 + r][lh * 16];
#pragma unroll
      for (int j = 0; j < 16; ++j) b.u[j] = p[j];
      acc[t] = __builtin_amdgcn_wmma_f32_16x16x32_bf16(
          false, a.bf, false, b.bf, (short)0, acc[t], false, false);
    }
    __syncthreads();
  }

  // C layout: vgpr rr -> row mi*16+rr (+8 for upper lane half), col = nj*16 + r
#pragma unroll
  for (int t = 0; t < 4; ++t)
#pragma unroll
    for (int rr = 0; rr < 8; ++rr)
      outs[mi * 16 + rr + lh * 8][(njbase + t) * 16 + r] = acc[t][rr];
  __syncthreads();

  // ------------- fused epilogue: bias (+ RMSNorm + RoPE for q/k) -------------
  const int rowi = tid >> 2;   // 0..63 token row within tile
  const int quad = tid & 3;    // 4 threads per row, 32 cols each
  const int gRow = m0 + rowi;
  const int bidx = gRow >> 11;         // / 2048
  const int ltok = gRow & (L_ - 1);

  float vals[32];
#pragma unroll
  for (int j = 0; j < 32; ++j)
    vals[j] = outs[rowi][quad * 32 + j] + bqkv[nbase + quad * 32 + j];

  if (sel == 2) {
    u16* dst = v_ws + ((size_t)(bidx * HEADS_ + head) * L_ + ltok) * HD_;
#pragma unroll
    for (int j = 0; j < 32; ++j) dst[quad * 32 + j] = f2bfu(vals[j]);
  } else {
    float ss = 0.0f;
#pragma unroll
    for (int j = 0; j < 32; ++j) ss += vals[j] * vals[j];
    ss += __shfl_xor(ss, 1, 32);
    ss += __shfl_xor(ss, 2, 32);
    const float rrms = rsqrtf(ss * (1.0f / 128.0f) + 1e-6f);
    // fold attention score scale 1/sqrt(128) into q here
    const float oscale = (sel == 0) ? 0.088388347648318447f : 1.0f;
    const float* scale = (sel == 0) ? q_scale : k_scale;
    u16* dst = ((sel == 0) ? q_ws : k_ws) +
               ((size_t)(bidx * HEADS_ + head) * L_ + ltok) * HD_;
#pragma unroll
    for (int p2 = 0; p2 < 16; ++p2) {
      const int d2 = quad * 16 + p2;                     // pair index 0..63
      const float t0 = vals[2 * p2]     * rrms * scale[2 * d2];
      const float t1 = vals[2 * p2 + 1] * rrms * scale[2 * d2 + 1];
      const float* pp = pe + (((size_t)ltok * 64 + d2) * 4);
      dst[2 * d2]     = f2bfu((pp[0] * t0 + pp[1] * t1) * oscale);
      dst[2 * d2 + 1] = f2bfu((pp[2] * t0 + pp[3] * t1) * oscale);
    }
  }
}

// ---------------------------------------------------------------------------
// Kernel 2: flash attention per (b, h, 128-row q tile). grid=(16,24,2)
// Each wave owns a 16-row band -> softmax reductions stay in-wave.
// ---------------------------------------------------------------------------
__global__ __launch_bounds__(256) void attention_kernel(
    const u16* __restrict__ q_ws, const u16* __restrict__ k_ws,
    const u16* __restrict__ v_ws, u16* __restrict__ ao)
{
  __shared__ u16 kl[64][136];   // K chunk [token][d]
  __shared__ u16 vl[128][72];   // V chunk transposed [d][token]
  __shared__ u16 pl[128][72];   // P tile bf16 [q-row][token]

  const int tid  = threadIdx.x;
  const int lane = tid & 31;
  const int wv   = tid >> 5;
  const int r    = lane & 15;
  const int lh   = lane >> 4;
  const int wband = wv * 16;

  const int b  = blockIdx.z, h = blockIdx.y;
  const int qb = blockIdx.x * 128;
  const size_t headoff = (size_t)(b * HEADS_ + h) * L_ * HD_;
  const u16* qp = q_ws + headoff + (size_t)qb * HD_;
  const u16* kp = k_ws + headoff;
  const u16* vp = v_ws + headoff;

  // Q fragments for this wave's 16 rows, K=128 -> 4 ksteps of 32
  FragBF qf[4];
#pragma unroll
  for (int ks = 0; ks < 4; ++ks) {
    const u16* g0 = qp + (size_t)(wband + r) * HD_ + ks * 32 + lh * 16;
    const u16* g1 = g0 + 8;
#pragma unroll
    for (int j = 0; j < 8; ++j) { qf[ks].u[j] = g0[j]; qf[ks].u[8 + j] = g1[j]; }
  }

  v8f vz = {};
  float mrow[8], lrow[8];
  v8f oacc[8];
#pragma unroll
  for (int i = 0; i < 8; ++i) { mrow[i] = -1e30f; lrow[i] = 0.0f; oacc[i] = vz; }

  for (int kc = 0; kc < L_; kc += 64) {
    // K tile: async global->LDS b128 (no VGPR round-trip on CDNA5)
#pragma unroll
    for (int it = 0; it < 4; ++it) {
      const int idx = (tid + it * 256) * 8;          // u16 index in 64x128
      const int tok = idx >> 7, d0 = idx & 127;
      async_ld16(&kl[tok][d0], kp + (size_t)(kc + tok) * HD_ + d0);
    }
    // V tile: load b128, transpose-scatter into vl[d][tok]
#pragma unroll
    for (int it = 0; it < 4; ++it) {
      const int idx = (tid + it * 256) * 8;
      const int tok = idx >> 7, d0 = idx & 127;
      uint4 tmp = *(const uint4*)(vp + (size_t)(kc + tok) * HD_ + d0);
      const u16* ts = (const u16*)&tmp;
#pragma unroll
      for (int j = 0; j < 8; ++j) vl[d0 + j][tok] = ts[j];
    }
    async_fence();
    __syncthreads();

    // S = Q K^T : 16 rows x 64 kv-cols (scale pre-folded into q)
    v8f sacc[4];
#pragma unroll
    for (int nj = 0; nj < 4; ++nj) {
      sacc[nj] = vz;
#pragma unroll
      for (int ks = 0; ks < 4; ++ks) {
        FragBF bfr;
        const u16* p = &kl[nj * 16 + r][ks * 32 + lh * 16];
#pragma unroll
        for (int j = 0; j < 16; ++j) bfr.u[j] = p[j];
        sacc[nj] = __builtin_amdgcn_wmma_f32_16x16x32_bf16(
            false, qf[ks].bf, false, bfr.bf, (short)0, sacc[nj], false, false);
      }
    }

    // online softmax; row = wband + rr + lh*8, cols across 16 lanes + 4 tiles
#pragma unroll
    for (int rr = 0; rr < 8; ++rr) {
      float mx = -1e30f;
#pragma unroll
      for (int nj = 0; nj < 4; ++nj) mx = fmaxf(mx, sacc[nj][rr]);
      for (int msk = 1; msk < 16; msk <<= 1) mx = fmaxf(mx, __shfl_xor(mx, msk, 32));
      const float mnew  = fmaxf(mrow[rr], mx);
      const float alpha = __expf(mrow[rr] - mnew);
      float rsum = 0.0f;
      const int row = wband + rr + lh * 8;
#pragma unroll
      for (int nj = 0; nj < 4; ++nj) {
        const float p = __expf(sacc[nj][rr] - mnew);
        rsum += p;
        pl[row][nj * 16 + r] = f2bfu(p);
      }
      for (int msk = 1; msk < 16; msk <<= 1) rsum += __shfl_xor(rsum, msk, 32);
      lrow[rr] = lrow[rr] * alpha + rsum;
      mrow[rr] = mnew;
#pragma unroll
      for (int nj = 0; nj < 8; ++nj) oacc[nj][rr] *= alpha;
    }
    __syncthreads();

    // O += P @ V  (K = 64 -> 2 ksteps of 32, N = 128 -> 8 tiles)
#pragma unroll
    for (int ks2 = 0; ks2 < 2; ++ks2) {
      FragBF pa;
      const u16* p0 = &pl[wband + r][ks2 * 32 + lh * 16];
      const u16* p1 = p0 + 8;
#pragma unroll
      for (int j = 0; j < 8; ++j) { pa.u[j] = p0[j]; pa.u[8 + j] = p1[j]; }
#pragma unroll
      for (int nj = 0; nj < 8; ++nj) {
        FragBF bfr;
        const u16* p = &vl[nj * 16 + r][ks2 * 32 + lh * 16];
#pragma unroll
        for (int j = 0; j < 16; ++j) bfr.u[j] = p[j];
        oacc[nj] = __builtin_amdgcn_wmma_f32_16x16x32_bf16(
            false, pa.bf, false, bfr.bf, (short)0, oacc[nj], false, false);
      }
    }
    __syncthreads();
  }

  // normalize and store to attn-out [b, l, h*d] as bf16
#pragma unroll
  for (int rr = 0; rr < 8; ++rr) {
    const float inv = 1.0f / lrow[rr];
    const int rowg = wband + rr + lh * 8;
    const size_t base = ((size_t)(b * L_ + qb + rowg)) * DIM_ + h * HD_;
#pragma unroll
    for (int nj = 0; nj < 8; ++nj)
      ao[base + nj * 16 + r] = f2bfu(oacc[nj][rr] * inv);
  }
}

// ---------------------------------------------------------------------------
// Kernel 3: out = attn @ Wproj^T + bproj (fp32 out). grid=(64, 24)
// ---------------------------------------------------------------------------
__global__ __launch_bounds__(256) void proj_kernel(
    const u16* __restrict__ ao, const float* __restrict__ Wproj,
    const float* __restrict__ bproj, float* __restrict__ out)
{
  __shared__ u16 as_[64][40];
  __shared__ u16 wsm[128][40];

  const int tid  = threadIdx.x;
  const int lane = tid & 31;
  const int wv   = tid >> 5;
  const int r    = lane & 15;
  const int lh   = lane >> 4;
  const int m0    = blockIdx.x * 64;
  const int nbase = blockIdx.y * 128;
  const int mi     = wv >> 1;
  const int njbase = (wv & 1) * 4;

  v8f vz = {};
  v8f acc[4];
  for (int t = 0; t < 4; ++t) acc[t] = vz;

  for (int kk = 0; kk < DIM_; kk += 32) {
    // A tile (already bf16): async global->LDS b128, one per thread
    {
      const int idx = tid * 8;
      const int rr = idx >> 5, cc = idx & 31;
      async_ld16(&as_[rr][cc], ao + (size_t)(m0 + rr) * DIM_ + kk + cc);
    }
    // W tile: f32 -> packed bf16
#pragma unroll
    for (int it = 0; it < 4; ++it) {
      const int idx = tid + it * 256;
      const int rr = idx >> 3, cc = (idx & 7) * 4;
      float4 v = *(const float4*)(Wproj + (size_t)(nbase + rr) * DIM_ + kk + cc);
      uint2 pk; pk.x = pk2bf(v.x, v.y); pk.y = pk2bf(v.z, v.w);
      *(uint2*)&wsm[rr][cc] = pk;
    }
    async_fence();
    __syncthreads();

    FragBF a;
    {
      const u16* p0 = &as_[mi * 16 + r][lh * 16];
      const u16* p1 = p0 + 8;
#pragma unroll
      for (int j = 0; j < 8; ++j) { a.u[j] = p0[j]; a.u[8 + j] = p1[j]; }
    }
#pragma unroll
    for (int t = 0; t < 4; ++t) {
      FragBF bfr;
      const u16* p = &wsm[(njbase + t) * 16 + r][lh * 16];
#pragma unroll
      for (int j = 0; j < 16; ++j) bfr.u[j] = p[j];
      acc[t] = __builtin_amdgcn_wmma_f32_16x16x32_bf16(
          false, a.bf, false, bfr.bf, (short)0, acc[t], false, false);
    }
    __syncthreads();
  }

#pragma unroll
  for (int t = 0; t < 4; ++t) {
    const int col = (njbase + t) * 16 + r;
    const float bias = bproj[nbase + col];
#pragma unroll
    for (int rr = 0; rr < 8; ++rr) {
      const int row = mi * 16 + rr + lh * 8;
      out[(size_t)(m0 + row) * DIM_ + nbase + col] = acc[t][rr] + bias;
    }
  }
}

extern "C" void kernel_launch(void* const* d_in, const int* in_sizes, int n_in,
                              void* d_out, int out_size, void* d_ws, size_t ws_size,
                              hipStream_t stream) {
  (void)in_sizes; (void)n_in; (void)out_size; (void)ws_size;
  const float* x    = (const float*)d_in[0];
  const float* pe   = (const float*)d_in[1];
  const float* Wqkv = (const float*)d_in[2];
  const float* bqkv = (const float*)d_in[3];
  const float* qs   = (const float*)d_in[4];
  const float* ks   = (const float*)d_in[5];
  const float* Wp   = (const float*)d_in[6];
  const float* bp   = (const float*)d_in[7];

  u16* q_ws = (u16*)d_ws;
  u16* k_ws = q_ws + (size_t)QH_;
  u16* v_ws = k_ws + (size_t)QH_;
  u16* ao   = v_ws + (size_t)QH_;
  float* out = (float*)d_out;

  qkv_fused_kernel<<<dim3(BL_ / 64, 72), 256, 0, stream>>>(
      x, pe, Wqkv, bqkv, qs, ks, q_ws, k_ws, v_ws);
  attention_kernel<<<dim3(L_ / 128, HEADS_, NB_), 256, 0, stream>>>(
      q_ws, k_ws, v_ws, ao);
  proj_kernel<<<dim3(BL_ / 64, DIM_ / 128), 256, 0, stream>>>(ao, Wp, bp, out);
}
